// Net_7129645711824
// MI455X (gfx1250) — matrix operand order repaired and checked by reading, not compiled, not run on hardware
//
#include <hip/hip_runtime.h>
#include <math.h>

#define N_NODES 50000
#define N_EDGES 1600000
#define F_IN    64
#define HID     32
#define EDIM    44
#define CGC     13
#define ZK      70      // 2*CGC + EDIM
#define ZKP     96      // padded to 3*32
#define BN_EPS  1e-5f

typedef __bf16 bf16;
typedef bf16  bf16x16 __attribute__((ext_vector_type(16)));
typedef float f32x8   __attribute__((ext_vector_type(8)));
typedef float f32x4   __attribute__((ext_vector_type(4)));

union FragU { bf16x16 v; f32x4 q[2]; };

// A fragment (16xK, bf16): contiguous 16 at k0
static __device__ __forceinline__ bf16x16 frag_a(const bf16* row, int k0) {
  FragU u;
  u.q[0] = *(const f32x4*)(row + k0);
  u.q[1] = *(const f32x4*)(row + k0 + 8);
  return u.v;
}
// B fragment (Kx16, bf16, weights stored transposed [n][k]): contiguous 16 at k0
static __device__ __forceinline__ bf16x16 frag_b(const bf16* row, int k0) {
  FragU u;
  u.q[0] = *(const f32x4*)(row + k0);
  u.q[1] = *(const f32x4*)(row + k0 + 8);
  return u.v;
}
static __device__ __forceinline__ f32x8 wmma_bf16(bf16x16 a, bf16x16 b, f32x8 c) {
  return __builtin_amdgcn_wmma_f32_16x16x32_bf16(false, a, false, b, (short)0, c,
                                                 false, false);
}

static __device__ __forceinline__ void atomAddF(float* p, float v) {
  unsafeAtomicAdd(p, v);   // global_atomic_add_f32
}
static __device__ __forceinline__ void atomMaxF(float* p, float v) {
  if (v >= 0.0f) atomicMax((int*)p, __float_as_int(v));
  else           atomicMin((unsigned int*)p, (unsigned int)__float_as_int(v));
}
// branchless numerically-stable softplus: v_exp + v_log only
static __device__ __forceinline__ float softplus_f(float g) {
  return fmaxf(g, 0.0f) + __logf(1.0f + __expf(-fabsf(g)));
}
// sigmoid with raw v_rcp_f32 (avoid IEEE div_scale/div_fixup expansion)
static __device__ __forceinline__ float sigmoid_f(float f) {
  return __builtin_amdgcn_rcpf(1.0f + __expf(-f));
}

// ---------------------------------------------------------------- utility
__global__ void k_zero(float* __restrict__ p, long n) {
  long i = blockIdx.x * (long)blockDim.x + threadIdx.x;
  if (i < n) p[i] = 0.0f;
}

__global__ void k_ce_init(const float* __restrict__ x, float* __restrict__ ce, int n) {
  int i = blockIdx.x * blockDim.x + threadIdx.x;
  if (i >= n) return;
  const float* xr = x + (long)i * F_IN;
  float* cr = ce + (long)i * CGC;
  #pragma unroll
  for (int c = 0; c < CGC; ++c) cr[c] = xr[c];
}

// ---------------------------------------------------------------- CGConv edges
// per 16-edge tile: z[16x96]bf16 @ (Wf|Ws)[96x16]bf16 via v_wmma_f32_16x16x32_bf16
__global__ void __launch_bounds__(128) k_cg(
    const float* __restrict__ x, const int* __restrict__ src,
    const int* __restrict__ dst, const float* __restrict__ ea,
    const float* __restrict__ Wf, const float* __restrict__ biasf,
    const float* __restrict__ Ws, const float* __restrict__ biass,
    float* __restrict__ ce, int nGroups) {
  __shared__ __align__(16) bf16 Wt[2][16][ZKP];       // weights, transposed+padded
  __shared__ __align__(16) bf16 zld[4][32][ZKP];      // z rows per wave
  __shared__ int dlds[4][32];

  const int tid  = threadIdx.x;
  const int wave = tid >> 5;
  const int lane = tid & 31;
  const int half = lane >> 4;
  const int l16  = lane & 15;

  for (int i = tid; i < 2 * 16 * ZKP; i += 128) {
    int mat = i / (16 * ZKP);
    int rem = i - mat * 16 * ZKP;
    int n = rem / ZKP, k = rem % ZKP;
    const float* W = mat ? Ws : Wf;
    float v = (n < CGC && k < ZK) ? W[k * CGC + n] : 0.0f;
    Wt[mat][n][k] = (bf16)v;
  }
  __syncthreads();

  bf16x16 bfr[3], bsr[3];
  #pragma unroll
  for (int kc = 0; kc < 3; ++kc) {
    int k0 = kc * 32 + 16 * half;
    bfr[kc] = frag_b(&Wt[0][l16][0], k0);
    bsr[kc] = frag_b(&Wt[1][l16][0], k0);
  }
  float cbf = (l16 < CGC) ? biasf[l16] : 0.0f;
  float cbs = (l16 < CGC) ? biass[l16] : 0.0f;

  for (int grp = blockIdx.x; grp < nGroups; grp += gridDim.x) {
    int e = grp * 128 + wave * 32 + lane;
    int s = src[e], d = dst[e];
    dlds[wave][lane] = d;
    bf16* zr = zld[wave][lane];
    const float* xd = x + (long)d * F_IN;
    const float* xs = x + (long)s * F_IN;
    #pragma unroll
    for (int j = 0; j < CGC; ++j) zr[j] = (bf16)xd[j];
    #pragma unroll
    for (int j = 0; j < CGC; ++j) zr[CGC + j] = (bf16)xs[j];
    const f32x4* ev = (const f32x4*)(ea + (long)e * EDIM);
    #pragma unroll
    for (int j = 0; j < EDIM / 4; ++j) {
      f32x4 q = ev[j];
      zr[2 * CGC + 4 * j + 0] = (bf16)q.x;
      zr[2 * CGC + 4 * j + 1] = (bf16)q.y;
      zr[2 * CGC + 4 * j + 2] = (bf16)q.z;
      zr[2 * CGC + 4 * j + 3] = (bf16)q.w;
    }
    #pragma unroll
    for (int j = ZK; j < ZKP; ++j) zr[j] = (bf16)0.0f;
    __syncthreads();

    #pragma unroll
    for (int t = 0; t < 2; ++t) {
      const bf16* ar = zld[wave][t * 16 + l16];
      f32x8 af = {0, 0, 0, 0, 0, 0, 0, 0};
      f32x8 ag = {0, 0, 0, 0, 0, 0, 0, 0};
      #pragma unroll
      for (int kc = 0; kc < 3; ++kc) {
        bf16x16 a = frag_a(ar, kc * 32 + 16 * half);
        af = wmma_bf16(a, bfr[kc], af);
        ag = wmma_bf16(a, bsr[kc], ag);
      }
      if (l16 < CGC) {
        #pragma unroll
        for (int r = 0; r < 8; ++r) {
          int row = t * 16 + r + 8 * half;
          int dd = dlds[wave][row];
          float msg = sigmoid_f(af[r] + cbf) * softplus_f(ag[r] + cbs);
          atomAddF(&ce[(long)dd * CGC + l16], msg);
        }
      }
    }
    __syncthreads();
  }
}

// ---------------------------------------------------------------- dense GEMM: out[n,32] = A[n,K] @ W[K,32]
__global__ void __launch_bounds__(128) k_gemm(
    const float* __restrict__ A, const float* __restrict__ W,
    float* __restrict__ out, int nRows, int K, int KP) {
  __shared__ __align__(16) bf16 Wt[32 * 64];
  __shared__ __align__(16) bf16 zld[4][16 * 64];
  const int tid = threadIdx.x, wave = tid >> 5, lane = tid & 31;
  const int half = lane >> 4, l16 = lane & 15;

  for (int i = tid; i < 32 * KP; i += 128) {
    int n = i / KP, k = i % KP;
    Wt[n * KP + k] = (bf16)((k < K) ? W[k * HID + n] : 0.0f);
  }
  int rowbase = blockIdx.x * 64 + wave * 16;
  {
    int m = rowbase + l16;
    bf16* zr = &zld[wave][l16 * KP];
    int j0 = half * (KP / 2);
    for (int j = 0; j < KP / 2; ++j) {
      int jj = j0 + j;
      float v = (m < nRows && jj < K) ? A[(long)m * K + jj] : 0.0f;
      zr[jj] = (bf16)v;
    }
  }
  __syncthreads();

  f32x8 acc0 = {0, 0, 0, 0, 0, 0, 0, 0};
  f32x8 acc1 = {0, 0, 0, 0, 0, 0, 0, 0};
  int kcnt = KP / 32;
  for (int kc = 0; kc < kcnt; ++kc) {
    bf16x16 a  = frag_a(&zld[wave][l16 * KP], kc * 32 + 16 * half);
    bf16x16 b0 = frag_b(&Wt[(l16) * KP],      kc * 32 + 16 * half);
    bf16x16 b1 = frag_b(&Wt[(16 + l16) * KP], kc * 32 + 16 * half);
    acc0 = wmma_bf16(a, b0, acc0);
    acc1 = wmma_bf16(a, b1, acc1);
  }
  #pragma unroll
  for (int r = 0; r < 8; ++r) {
    int m = rowbase + r + 8 * half;
    if (m < nRows) {
      out[(long)m * HID + l16]      = acc0[r];
      out[(long)m * HID + 16 + l16] = acc1[r];
    }
  }
}

// ---------------------------------------------------------------- GAT pieces
__global__ void k_dot_init(const float* __restrict__ h, const float* __restrict__ as_,
                           const float* __restrict__ ad_, float* __restrict__ a_src,
                           float* __restrict__ a_dst, float* __restrict__ mx,
                           float* __restrict__ den, int n) {
  int i = blockIdx.x * blockDim.x + threadIdx.x;
  if (i >= n) return;
  const float* hr = h + (long)i * HID;
  float s = 0.0f, d = 0.0f;
  #pragma unroll
  for (int c = 0; c < HID; ++c) { float v = hr[c]; s += v * as_[c]; d += v * ad_[c]; }
  a_src[i] = s; a_dst[i] = d;
  mx[i] = -INFINITY; den[i] = 0.0f;
}

// pass 1: alpha = leaky_relu(a_src[s]+a_dst[d]); stash alpha; segment max
__global__ void k_edge_max(const int* __restrict__ src, const int* __restrict__ dst,
                           const float* __restrict__ a_src, const float* __restrict__ a_dst,
                           float* __restrict__ mx, float* __restrict__ alpha_buf, int tot) {
  int gid = blockIdx.x * blockDim.x + threadIdx.x;
  if (gid >= tot) return;
  int s, d;
  if (gid < N_EDGES) { s = src[gid]; d = dst[gid]; } else { s = d = gid - N_EDGES; }
  float a = a_src[s] + a_dst[d];
  a = (a > 0.0f) ? a : 0.2f * a;          // leaky_relu 0.2
  alpha_buf[gid] = a;
  atomMaxF(&mx[d], a);
}

// pass 2: e = exp(alpha - mx[d]); stash e (overwrite alpha slot); segment sum
__global__ void k_edge_den(const int* __restrict__ dst, const float* __restrict__ mx,
                           float* __restrict__ alpha_buf, float* __restrict__ den, int tot) {
  int gid = blockIdx.x * blockDim.x + threadIdx.x;
  if (gid >= tot) return;
  int d = (gid < N_EDGES) ? dst[gid] : gid - N_EDGES;
  float e = __expf(alpha_buf[gid] - mx[d]);
  alpha_buf[gid] = e;
  atomAddF(&den[d], e);
}

// pass 3: coef = e/den[d]; acc[d] += coef * h[s]
__global__ void k_edge_scatter(const int* __restrict__ src, const int* __restrict__ dst,
                               const float* __restrict__ alpha_buf,
                               const float* __restrict__ den,
                               const float* __restrict__ h, float* __restrict__ acc, int tot) {
  int gid = blockIdx.x * blockDim.x + threadIdx.x;
  if (gid >= tot) return;
  int s, d;
  if (gid < N_EDGES) { s = src[gid]; d = dst[gid]; } else { s = d = gid - N_EDGES; }
  float coef = alpha_buf[gid] * __builtin_amdgcn_rcpf(den[d]);
  const float* hs = h + (long)s * HID;
  float* ad = acc + (long)d * HID;
  #pragma unroll
  for (int c = 0; c < HID; ++c) atomAddF(&ad[c], coef * hs[c]);
}

// ---------------------------------------------------------------- GCN pieces
__global__ void k_deg_init(float* __restrict__ deg, int n) {
  int i = blockIdx.x * blockDim.x + threadIdx.x;
  if (i < n) deg[i] = 1.0f;   // self-loop
}
__global__ void k_deg_acc(const int* __restrict__ dst, float* __restrict__ deg, int e) {
  int i = blockIdx.x * blockDim.x + threadIdx.x;
  if (i < e) atomAddF(&deg[dst[i]], 1.0f);
}
__global__ void k_gcn_scatter(const int* __restrict__ src, const int* __restrict__ dst,
                              const float* __restrict__ deg, const float* __restrict__ h,
                              float* __restrict__ acc, int tot) {
  int gid = blockIdx.x * blockDim.x + threadIdx.x;
  if (gid >= tot) return;
  int s, d;
  if (gid < N_EDGES) { s = src[gid]; d = dst[gid]; } else { s = d = gid - N_EDGES; }
  float norm = rsqrtf(deg[s]) * rsqrtf(deg[d]);
  const float* hs = h + (long)s * HID;
  float* ad = acc + (long)d * HID;
  #pragma unroll
  for (int c = 0; c < HID; ++c) atomAddF(&ad[c], norm * hs[c]);
}

// ---------------------------------------------------------------- BatchNorm (training-mode biased stats)
__global__ void k_bn_stats(const float* __restrict__ in, const float* __restrict__ bias,
                           int C, int n, float* __restrict__ gsums) {
  __shared__ float ssum[HID], ssq[HID];
  int t = threadIdx.x;
  if (t < C) { ssum[t] = 0.0f; ssq[t] = 0.0f; }
  __syncthreads();
  int i = blockIdx.x * blockDim.x + t;
  if (i < n) {
    const float* r = in + (long)i * C;
    for (int c = 0; c < C; ++c) {
      float v = r[c] + (bias ? bias[c] : 0.0f);
      v = fmaxf(v, 0.0f);                       // relu before BN
      atomicAdd(&ssum[c], v);
      atomicAdd(&ssq[c], v * v);
    }
  }
  __syncthreads();
  if (t < C) { atomAddF(&gsums[t], ssum[t]); atomAddF(&gsums[32 + t], ssq[t]); }
}

__global__ void k_bn_apply(const float* __restrict__ in, float* __restrict__ out,
                           const float* __restrict__ g, const float* __restrict__ b,
                           const float* __restrict__ bias, const float* __restrict__ gsums,
                           int C, int n) {
  int i = blockIdx.x * blockDim.x + threadIdx.x;
  if (i >= n) return;
  const float* r = in + (long)i * C;
  float* o = out + (long)i * C;
  float inv_n = 1.0f / (float)n;
  for (int c = 0; c < C; ++c) {
    float mean = gsums[c] * inv_n;
    float var  = gsums[32 + c] * inv_n - mean * mean;
    float v = fmaxf(r[c] + (bias ? bias[c] : 0.0f), 0.0f);
    o[c] = (v - mean) * rsqrtf(var + BN_EPS) * g[c] + b[c];
  }
}

// ---------------------------------------------------------------- fused head: (h@W00+b00)@W0+b0 == h@(W00@W0)+const
__global__ void k_fuse(const float* __restrict__ W00, const float* __restrict__ b00,
                       const float* __restrict__ W0, const float* __restrict__ b0,
                       float* __restrict__ wf) {
  int j = threadIdx.x;
  if (j < 141) {
    float s = 0.0f;
    for (int c = 0; c < HID; ++c) s += W00[j * HID + c] * W0[c];
    wf[j] = s;
  } else if (j == 141) {
    float s = b0[0];
    for (int c = 0; c < HID; ++c) s += b00[c] * W0[c];
    wf[141] = s;
  }
}

__global__ void k_final(const float* __restrict__ x1, const float* __restrict__ x2,
                        const float* __restrict__ x3, const float* __restrict__ xe,
                        const float* __restrict__ x6, const float* __restrict__ wf,
                        float* __restrict__ out, int n) {
  int i = blockIdx.x * blockDim.x + threadIdx.x;
  if (i >= n) return;
  float s = wf[141];
  const float* r1 = x1 + (long)i * HID;
  const float* r2 = x2 + (long)i * HID;
  const float* r3 = x3 + (long)i * HID;
  const float* re = xe + (long)i * CGC;
  const float* r6 = x6 + (long)i * HID;
  #pragma unroll
  for (int c = 0; c < HID; ++c) s += r1[c] * wf[c];
  #pragma unroll
  for (int c = 0; c < HID; ++c) s += r2[c] * wf[32 + c];
  #pragma unroll
  for (int c = 0; c < HID; ++c) s += r3[c] * wf[64 + c];
  #pragma unroll
  for (int c = 0; c < CGC; ++c) s += re[c] * wf[96 + c];
  #pragma unroll
  for (int c = 0; c < HID; ++c) s += r6[c] * wf[109 + c];
  out[i] = fmaxf(s, 0.0f);
}

// ---------------------------------------------------------------- host
static inline int nb(long n, int b) { return (int)((n + b - 1) / b); }

extern "C" void kernel_launch(void* const* d_in, const int* in_sizes, int n_in,
                              void* d_out, int out_size, void* d_ws, size_t ws_size,
                              hipStream_t stream) {
  // setup_inputs() insertion order, params flattened depth-first in insertion order
  const float* x    = (const float*)d_in[0];
  const int*   adj  = (const int*)d_in[1];
  const float* ea   = (const float*)d_in[2];
  const float* g1W  = (const float*)d_in[3];
  const float* g1as = (const float*)d_in[4];
  const float* g1ad = (const float*)d_in[5];
  const float* g1b  = (const float*)d_in[6];
  const float* g2W  = (const float*)d_in[7];
  const float* g2as = (const float*)d_in[8];
  const float* g2ad = (const float*)d_in[9];
  const float* g2b  = (const float*)d_in[10];
  const float* g3W  = (const float*)d_in[11];
  const float* g3as = (const float*)d_in[12];
  const float* g3ad = (const float*)d_in[13];
  const float* g3b  = (const float*)d_in[14];
  const float* cgWf = (const float*)d_in[15];
  const float* cgbf = (const float*)d_in[16];
  const float* cgWs = (const float*)d_in[17];
  const float* cgbs = (const float*)d_in[18];
  const float* gcW  = (const float*)d_in[19];
  const float* gcb  = (const float*)d_in[20];
  const float* bn1g = (const float*)d_in[21];
  const float* bn1b = (const float*)d_in[22];
  const float* bn2g = (const float*)d_in[23];
  const float* bn2b = (const float*)d_in[24];
  const float* bn3g = (const float*)d_in[25];
  const float* bn3b = (const float*)d_in[26];
  const float* bn4g = (const float*)d_in[27];
  const float* bn4b = (const float*)d_in[28];
  const float* bn6g = (const float*)d_in[29];
  const float* bn6b = (const float*)d_in[30];
  const float* f00W = (const float*)d_in[31];
  const float* f00b = (const float*)d_in[32];
  const float* f0W  = (const float*)d_in[33];
  const float* f0b  = (const float*)d_in[34];

  const int* src = adj;
  const int* dst = adj + N_EDGES;

  float* ws = (float*)d_ws;
  const long NH = (long)N_NODES * HID;
  float* h_buf = ws;                 // N*32
  float* acc   = ws + NH;            // N*32
  float* x1    = ws + 2 * NH;        // N*32
  float* x2    = ws + 3 * NH;        // N*32
  float* x3    = ws + 4 * NH;        // N*32
  float* x6    = ws + 5 * NH;        // N*32
  float* ce    = ws + 6 * NH;                         // N*13
  float* xe    = ce + (long)N_NODES * CGC;            // N*13
  float* a_src = xe + (long)N_NODES * CGC;            // N
  float* a_dst = a_src + N_NODES;                     // N
  float* mx    = a_dst + N_NODES;                     // N
  float* den   = mx + N_NODES;                        // N
  float* deg   = den + N_NODES;                       // N
  float* bns   = deg + N_NODES;                       // 64 (sum[32], sumsq[32])
  float* wf    = bns + 64;                            // 142 fused head weights
  float* alpha = wf + 144;                            // E+N per-edge scratch

  const int TOT = N_EDGES + N_NODES;   // edges + self-loops

  // ---- CGConv branch: ce = x[:, :13] + scatter(msg); xe = bn4(relu(ce))
  k_ce_init<<<nb(N_NODES, 256), 256, 0, stream>>>(x, ce, N_NODES);
  k_cg<<<2048, 128, 0, stream>>>(x, src, dst, ea, cgWf, cgbf, cgWs, cgbs, ce,
                                 N_EDGES / 128);
  k_zero<<<1, 64, 0, stream>>>(bns, 64);
  k_bn_stats<<<nb(N_NODES, 256), 256, 0, stream>>>(ce, nullptr, CGC, N_NODES, bns);
  k_bn_apply<<<nb(N_NODES, 256), 256, 0, stream>>>(ce, xe, bn4g, bn4b, nullptr, bns,
                                                   CGC, N_NODES);

  // ---- degrees for GCN (independent of everything else)
  k_deg_init<<<nb(N_NODES, 256), 256, 0, stream>>>(deg, N_NODES);
  k_deg_acc<<<nb(N_EDGES, 256), 256, 0, stream>>>(dst, deg, N_EDGES);

  // ---- three GAT layers
  struct GatP { const float* in; int K, KP; const float* W, *as, *ad, *b, *g, *bb; float* out; };
  const GatP gats[3] = {
    { x,  F_IN, 64, g1W, g1as, g1ad, g1b, bn1g, bn1b, x1 },
    { x1, HID,  32, g2W, g2as, g2ad, g2b, bn2g, bn2b, x2 },
    { x2, HID,  32, g3W, g3as, g3ad, g3b, bn3g, bn3b, x3 },
  };
  for (int L = 0; L < 3; ++L) {
    const GatP& p = gats[L];
    k_gemm<<<nb(N_NODES, 64), 128, 0, stream>>>(p.in, p.W, h_buf, N_NODES, p.K, p.KP);
    k_dot_init<<<nb(N_NODES, 256), 256, 0, stream>>>(h_buf, p.as, p.ad, a_src, a_dst,
                                                     mx, den, N_NODES);
    k_zero<<<nb(NH, 256), 256, 0, stream>>>(acc, NH);
    k_edge_max<<<nb(TOT, 256), 256, 0, stream>>>(src, dst, a_src, a_dst, mx, alpha, TOT);
    k_edge_den<<<nb(TOT, 256), 256, 0, stream>>>(dst, mx, alpha, den, TOT);
    k_edge_scatter<<<nb(TOT, 256), 256, 0, stream>>>(src, dst, alpha, den, h_buf, acc,
                                                     TOT);
    k_zero<<<1, 64, 0, stream>>>(bns, 64);
    k_bn_stats<<<nb(N_NODES, 256), 256, 0, stream>>>(acc, p.b, HID, N_NODES, bns);
    k_bn_apply<<<nb(N_NODES, 256), 256, 0, stream>>>(acc, p.out, p.g, p.bb, p.b, bns,
                                                     HID, N_NODES);
  }

  // ---- GCN branch: x6 = bn6(relu(gcn(xe)))
  k_gemm<<<nb(N_NODES, 64), 128, 0, stream>>>(xe, gcW, h_buf, N_NODES, CGC, 32);
  k_zero<<<nb(NH, 256), 256, 0, stream>>>(acc, NH);
  k_gcn_scatter<<<nb(TOT, 256), 256, 0, stream>>>(src, dst, deg, h_buf, acc, TOT);
  k_zero<<<1, 64, 0, stream>>>(bns, 64);
  k_bn_stats<<<nb(N_NODES, 256), 256, 0, stream>>>(acc, gcb, HID, N_NODES, bns);
  k_bn_apply<<<nb(N_NODES, 256), 256, 0, stream>>>(acc, x6, bn6g, bn6b, gcb, bns,
                                                   HID, N_NODES);

  // ---- fused linear head (no nonlinearity between f00 and f0)
  k_fuse<<<1, 192, 0, stream>>>(f00W, f00b, f0W, f0b, wf);
  k_final<<<nb(N_NODES, 256), 256, 0, stream>>>(x1, x2, x3, xe, x6, wf,
                                                (float*)d_out, N_NODES);
}
